// CFATextAdapter_79637283603290
// MI455X (gfx1250) — matrix operand order repaired and checked by reading, not compiled, not run on hardware
//
#include <hip/hip_runtime.h>
#include <hip/hip_bf16.h>
#include <math.h>

// ---------------------------------------------------------------------------
// CDNA5 (gfx1250) implementation of the CFA text adapter.
// All heavy GEMMs run through V_WMMA_F32_16X16X32_F16 (f16 inputs, f32 acc).
// Attention is a fused flash-style kernel (online softmax) so the (B,H,T,S)
// score tensor never hits HBM. The broadcast structure of K_base/V_base is
// exploited: text-side k/v projections are computed once (1000 rows) instead
// of per-batch (65024 rows).
// Workspace requirement: ~286 MB (71,292,928 floats).
// ---------------------------------------------------------------------------

typedef __attribute__((ext_vector_type(16))) _Float16 v16h;
typedef __attribute__((ext_vector_type(8)))  _Float16 v8h;
typedef __attribute__((ext_vector_type(8)))  float    v8f;

#define DV   768
#define DTXT 512
#define NH   8
#define HD   64
#define PFX  16
#define RRK  16
#define BNK  256
#define BBATCH 64
#define TT   196
#define CC   1000
#define NVROWS (BBATCH*TT)   /* 12544 */
#define STOT (PFX + CC)      /* 1016  */

// ---------------------------------------------------------------------------
// WMMA helpers. Layouts per CDNA5 ISA 7.12.2:
//  A 16x32 f16: lane L (r=L&15, hi=L>>4): halves 0..7 = A[r][hi*8+j],
//               halves 8..15 = A[r][16+hi*8+j]
//  B 32x16 f16: lane L (n=L&15, hi=L>>4): halves 0..7 = B[hi*8+j][n],
//               halves 8..15 = B[16+hi*8+j][n]
//  C 16x16 f32: VGPR i: row = i + 8*hi, col = lane&15
// ---------------------------------------------------------------------------
__device__ __forceinline__ v8f wmma16(v16h a, v16h b, v8f c) {
  return __builtin_amdgcn_wmma_f32_16x16x32_f16(false, a, false, b, (short)0, c,
                                                false, false);
}
__device__ __forceinline__ v16h frag_a(const _Float16* p, int hi) {
  union { v16h v; v8h h[2]; } u;
  u.h[0] = *reinterpret_cast<const v8h*>(p + hi * 8);
  u.h[1] = *reinterpret_cast<const v8h*>(p + 16 + hi * 8);
  return u.v;
}
__device__ __forceinline__ v16h frag_b(const _Float16* p, int hi) {
  union { v16h v; v8h h[2]; } u;
  u.h[0] = *reinterpret_cast<const v8h*>(p + hi * 8);
  u.h[1] = *reinterpret_cast<const v8h*>(p + 16 + hi * 8);
  return u.v;
}
__device__ __forceinline__ void cvt4(_Float16* d, float4 f) {
  d[0] = (_Float16)f.x; d[1] = (_Float16)f.y;
  d[2] = (_Float16)f.z; d[3] = (_Float16)f.w;
}

// ---------------------------------------------------------------------------
// Generic WMMA GEMM:  C[M,N] = act( alpha * (A @ B') + bias )
//   trans_b=1: B is (N,K) row-major (weights, A @ B^T)
//   trans_b=0: B is (K_B,N) row-major (A @ B); rows >= K_B read as zero
//   K must be a multiple of 32. epilogue: 0 none, 1 qgelu, 2 sigmoid.
// Tile: 128x128x32, 256 threads (8 waves), wave grid 4x2, each wave 32x64.
// ---------------------------------------------------------------------------
__global__ __launch_bounds__(256) void gemm_wmma(
    const float* __restrict__ A, const float* __restrict__ B,
    const float* __restrict__ bias, float* __restrict__ C,
    int M, int N, int K, int trans_b, int K_B, int ldc,
    int epilogue, const float* __restrict__ alpha_ptr, int alpha_mode)
{
  __shared__ __align__(16) _Float16 sA[128 * 48];
  __shared__ __align__(16) _Float16 sB[128 * 48];
  const int tid  = threadIdx.x;
  const int lane = tid & 31, wave = tid >> 5;
  const int wm = wave >> 1, wn = wave & 1;
  const int m0 = blockIdx.y * 128, n0 = blockIdx.x * 128;
  const int r = lane & 15, hi = lane >> 4;

  const v8f vzero = {};
  v8f acc[2][4];
  for (int i = 0; i < 2; ++i)
    for (int j = 0; j < 4; ++j) acc[i][j] = vzero;

  for (int k0 = 0; k0 < K; k0 += 32) {
    {  // stage A tile (128 x 32) fp32 -> f16
      int row = tid >> 1, cb = (tid & 1) * 16;
      int gr = m0 + row;
      _Float16* d = &sA[row * 48 + cb];
      if (gr < M) {
        const float4* s4 =
            reinterpret_cast<const float4*>(A + (size_t)gr * K + k0 + cb);
        cvt4(d + 0, s4[0]); cvt4(d + 4, s4[1]);
        cvt4(d + 8, s4[2]); cvt4(d + 12, s4[3]);
      } else {
        for (int j = 0; j < 16; ++j) d[j] = (_Float16)0.f;
      }
    }
    if (trans_b) {  // stage B tile from (N,K): rows are output cols
      int row = tid >> 1, cb = (tid & 1) * 16;
      int gn = n0 + row;
      _Float16* d = &sB[row * 48 + cb];
      if (gn < N) {
        const float4* s4 =
            reinterpret_cast<const float4*>(B + (size_t)gn * K + k0 + cb);
        cvt4(d + 0, s4[0]); cvt4(d + 4, s4[1]);
        cvt4(d + 8, s4[2]); cvt4(d + 12, s4[3]);
      } else {
        for (int j = 0; j < 16; ++j) d[j] = (_Float16)0.f;
      }
    } else {  // stage B tile from (K_B,N) with transpose into sB[n][k]
      int kk = tid & 31, nb = (tid >> 5) * 16;
      int gk = k0 + kk;
      for (int j = 0; j < 16; ++j) {
        int gn = n0 + nb + j;
        float f = (gk < K_B && gn < N) ? B[(size_t)gk * N + gn] : 0.f;
        sB[(nb + j) * 48 + kk] = (_Float16)f;
      }
    }
    __syncthreads();
    v16h af[2], bfr[4];
    for (int mi = 0; mi < 2; ++mi)
      af[mi] = frag_a(&sA[(wm * 32 + mi * 16 + r) * 48], hi);
    for (int ni = 0; ni < 4; ++ni)
      bfr[ni] = frag_b(&sB[(wn * 64 + ni * 16 + r) * 48], hi);
    for (int mi = 0; mi < 2; ++mi)
      for (int ni = 0; ni < 4; ++ni)
        acc[mi][ni] = wmma16(af[mi], bfr[ni], acc[mi][ni]);
    __syncthreads();
  }

  float alpha = 1.f;
  if (alpha_ptr) {
    float a = *alpha_ptr;
    alpha = alpha_mode ? expf(fminf(a, 4.6051701859880914f)) : a;
  }
  for (int mi = 0; mi < 2; ++mi)
    for (int ni = 0; ni < 4; ++ni) {
      int col = n0 + wn * 64 + ni * 16 + r;
      if (col >= N) continue;
      float bv = bias ? bias[col] : 0.f;
      for (int i = 0; i < 8; ++i) {
        int row = m0 + wm * 32 + mi * 16 + hi * 8 + i;
        if (row >= M) continue;
        float v = acc[mi][ni][i] * alpha + bv;
        if (epilogue == 1)      v = v / (1.f + expf(-1.702f * v));  // qgelu
        else if (epilogue == 2) v = 1.f / (1.f + expf(-v));         // sigmoid
        C[(size_t)row * ldc + col] = v;
      }
    }
}

// ---------------------------------------------------------------------------
// Fused flash-style multi-head cross attention.
// grid = (B*H, ceil(T/32)); block = 256 (8 waves).
// K/V sources: rows [0,16) from per-batch prefix, rows [16,1016) from shared
// text projections. Pass A: online row max/sum via WMMA scores. Pass B:
// recompute scores, normalize to f16 probs in LDS, WMMA P@V.
// ---------------------------------------------------------------------------
__global__ __launch_bounds__(256) void mha_kernel(
    const float* __restrict__ qp,
    const float* __restrict__ kpre, const float* __restrict__ vpre,
    const float* __restrict__ ktext, const float* __restrict__ vtext,
    float* __restrict__ out)
{
  __shared__ __align__(16) _Float16 sQ[32 * 72];
  __shared__ __align__(16) _Float16 sK[128 * 72];
  __shared__ __align__(16) _Float16 sV[64 * 136];   // transposed: [d][s]
  __shared__ __align__(16) float    sS[32 * 130];   // scores; reused as f16 P
  __shared__ float rowM[32], rowL[32], red[32 * 8];

  const int tid  = threadIdx.x;
  const int lane = tid & 31, wave = tid >> 5;
  const int b = blockIdx.x >> 3, h = blockIdx.x & 7;
  const int t0 = blockIdx.y * 32;
  const int r = lane & 15, hi = lane >> 4;

  {  // stage Q tile once (32 x 64)
    int row = tid >> 3, cb = (tid & 7) * 8;
    int t = t0 + row;
    _Float16* d = &sQ[row * 72 + cb];
    if (t < TT) {
      const float4* s4 = reinterpret_cast<const float4*>(
          qp + ((size_t)(b * TT + t)) * DV + h * HD + cb);
      cvt4(d + 0, s4[0]); cvt4(d + 4, s4[1]);
    } else {
      for (int j = 0; j < 8; ++j) d[j] = (_Float16)0.f;
    }
  }
  if (tid < 32) { rowM[tid] = -3.0e38f; rowL[tid] = 0.f; }
  __syncthreads();

  // ---------------- pass A: row statistics ----------------
  for (int ch = 0; ch < 8; ++ch) {
    const int s0 = ch * 128;
    {  // stage K chunk (128 x 64)
      int row = tid >> 1, cb = (tid & 1) * 32;
      int s = s0 + row;
      const float* src = nullptr;
      if (s < PFX)       src = kpre + ((size_t)(b * PFX + s)) * DV + h * HD;
      else if (s < STOT) src = ktext + ((size_t)(s - PFX)) * DV + h * HD;
      _Float16* d = &sK[row * 72 + cb];
      if (src) {
        const float4* s4 = reinterpret_cast<const float4*>(src + cb);
        for (int jj = 0; jj < 8; ++jj) cvt4(d + jj * 4, s4[jj]);
      } else {
        for (int j = 0; j < 32; ++j) d[j] = (_Float16)0.f;
      }
    }
    __syncthreads();
    {  // scores: each wave one 16-col tile, 2 row tiles, K=64
      v8f a0 = {}, a1 = {};
      for (int kt = 0; kt < 2; ++kt) {
        v16h qa0 = frag_a(&sQ[(r) * 72 + kt * 32], hi);
        v16h qa1 = frag_a(&sQ[(16 + r) * 72 + kt * 32], hi);
        v16h kb  = frag_b(&sK[(wave * 16 + r) * 72 + kt * 32], hi);
        a0 = wmma16(qa0, kb, a0);
        a1 = wmma16(qa1, kb, a1);
      }
      for (int i = 0; i < 8; ++i) {
        sS[(hi * 8 + i) * 130 + wave * 16 + r]        = a0[i] * 0.125f;
        sS[(16 + hi * 8 + i) * 130 + wave * 16 + r]   = a1[i] * 0.125f;
      }
    }
    __syncthreads();
    {  // online max / sum, 8 threads per row
      int srow = tid >> 3, sub = tid & 7;
      float mloc = -3.0e38f;
      for (int c = sub; c < 128; c += 8)
        if (s0 + c < STOT) mloc = fmaxf(mloc, sS[srow * 130 + c]);
      red[srow * 8 + sub] = mloc;
      __syncthreads();
      float cmax = red[srow * 8];
      for (int j = 1; j < 8; ++j) cmax = fmaxf(cmax, red[srow * 8 + j]);
      float mold = rowM[srow];
      float mnew = fmaxf(mold, cmax);
      float sloc = 0.f;
      for (int c = sub; c < 128; c += 8)
        if (s0 + c < STOT) sloc += expf(sS[srow * 130 + c] - mnew);
      __syncthreads();
      red[srow * 8 + sub] = sloc;
      __syncthreads();
      if (sub == 0) {
        float csum = 0.f;
        for (int j = 0; j < 8; ++j) csum += red[srow * 8 + j];
        rowL[srow] = rowL[srow] * expf(mold - mnew) + csum;
        rowM[srow] = mnew;
      }
      __syncthreads();
    }
  }

  // ---------------- pass B: normalized probs @ V ----------------
  v8f accO = {};
  const int wm = wave >> 2, wn = wave & 3;
  for (int ch = 0; ch < 8; ++ch) {
    const int s0 = ch * 128;
    __syncthreads();
    {  // stage K chunk
      int row = tid >> 1, cb = (tid & 1) * 32;
      int s = s0 + row;
      const float* src = nullptr;
      if (s < PFX)       src = kpre + ((size_t)(b * PFX + s)) * DV + h * HD;
      else if (s < STOT) src = ktext + ((size_t)(s - PFX)) * DV + h * HD;
      _Float16* d = &sK[row * 72 + cb];
      if (src) {
        const float4* s4 = reinterpret_cast<const float4*>(src + cb);
        for (int jj = 0; jj < 8; ++jj) cvt4(d + jj * 4, s4[jj]);
      } else {
        for (int j = 0; j < 32; ++j) d[j] = (_Float16)0.f;
      }
    }
    {  // stage V chunk transposed: sV[d][s]
      int row = tid >> 1, cb = (tid & 1) * 32;
      int s = s0 + row;
      const float* src = nullptr;
      if (s < PFX)       src = vpre + ((size_t)(b * PFX + s)) * DV + h * HD;
      else if (s < STOT) src = vtext + ((size_t)(s - PFX)) * DV + h * HD;
      if (src) {
        const float4* s4 = reinterpret_cast<const float4*>(src + cb);
        for (int jj = 0; jj < 8; ++jj) {
          float4 f = s4[jj];
          sV[(cb + jj * 4 + 0) * 136 + row] = (_Float16)f.x;
          sV[(cb + jj * 4 + 1) * 136 + row] = (_Float16)f.y;
          sV[(cb + jj * 4 + 2) * 136 + row] = (_Float16)f.z;
          sV[(cb + jj * 4 + 3) * 136 + row] = (_Float16)f.w;
        }
      } else {
        for (int j = 0; j < 32; ++j) sV[(cb + j) * 136 + row] = (_Float16)0.f;
      }
    }
    __syncthreads();
    {  // recompute scores
      v8f a0 = {}, a1 = {};
      for (int kt = 0; kt < 2; ++kt) {
        v16h qa0 = frag_a(&sQ[(r) * 72 + kt * 32], hi);
        v16h qa1 = frag_a(&sQ[(16 + r) * 72 + kt * 32], hi);
        v16h kb  = frag_b(&sK[(wave * 16 + r) * 72 + kt * 32], hi);
        a0 = wmma16(qa0, kb, a0);
        a1 = wmma16(qa1, kb, a1);
      }
      for (int i = 0; i < 8; ++i) {
        sS[(hi * 8 + i) * 130 + wave * 16 + r]      = a0[i] * 0.125f;
        sS[(16 + hi * 8 + i) * 130 + wave * 16 + r] = a1[i] * 0.125f;
      }
    }
    __syncthreads();
    {  // normalize -> f16 probs overlaid on sS (register-staged for aliasing)
      int srow = tid >> 3, sub = tid & 7;
      float mrow = rowM[srow];
      float inv = 1.f / rowL[srow];
      float pr[16];
      int cnt = 0;
      for (int c = sub; c < 128; c += 8) {
        float p = 0.f;
        if (s0 + c < STOT) p = expf(sS[srow * 130 + c] - mrow) * inv;
        pr[cnt++] = p;
      }
      __syncthreads();
      _Float16* sPh = reinterpret_cast<_Float16*>(sS);
      cnt = 0;
      for (int c = sub; c < 128; c += 8)
        sPh[srow * 128 + c] = (_Float16)pr[cnt++];
    }
    __syncthreads();
    {  // P (32x128) @ V (128x64): each wave one 16x16 output tile
      const _Float16* sPh = reinterpret_cast<const _Float16*>(sS);
      for (int kt = 0; kt < 4; ++kt) {
        v16h pa = frag_a(&sPh[(wm * 16 + r) * 128 + kt * 32], hi);
        v16h vb = frag_b(&sV[(wn * 16 + r) * 136 + kt * 32], hi);
        accO = wmma16(pa, vb, accO);
      }
    }
  }

  {  // write output tile
    int col = h * HD + wn * 16 + r;
    for (int i = 0; i < 8; ++i) {
      int t = t0 + wm * 16 + hi * 8 + i;
      if (t < TT)
        out[((size_t)(b * TT + t)) * DV + col] = accO[i];
    }
  }
}

// ---------------------------------------------------------------------------
// Small elementwise / reduction kernels
// ---------------------------------------------------------------------------
__global__ __launch_bounds__(256) void ln_kernel(
    const float* __restrict__ in, float* __restrict__ out,
    const float* __restrict__ g, const float* __restrict__ bb, int D)
{
  __shared__ float red[256];
  int row = blockIdx.x, tid = threadIdx.x;
  const float* x = in + (size_t)row * D;
  float s = 0.f;
  for (int i = tid; i < D; i += 256) s += x[i];
  red[tid] = s; __syncthreads();
  for (int o = 128; o > 0; o >>= 1) { if (tid < o) red[tid] += red[tid + o]; __syncthreads(); }
  float mean = red[0] / D;
  __syncthreads();
  float v = 0.f;
  for (int i = tid; i < D; i += 256) { float d = x[i] - mean; v += d * d; }
  red[tid] = v; __syncthreads();
  for (int o = 128; o > 0; o >>= 1) { if (tid < o) red[tid] += red[tid + o]; __syncthreads(); }
  float rstd = rsqrtf(red[0] / D + 1e-5f);
  float* y = out + (size_t)row * D;
  for (int i = tid; i < D; i += 256) y[i] = (x[i] - mean) * rstd * g[i] + bb[i];
}

__global__ __launch_bounds__(256) void l2norm_kernel(
    const float* __restrict__ in, float* __restrict__ out, int D)
{
  __shared__ float red[256];
  int row = blockIdx.x, tid = threadIdx.x;
  const float* x = in + (size_t)row * D;
  float s = 0.f;
  for (int i = tid; i < D; i += 256) s += x[i] * x[i];
  red[tid] = s; __syncthreads();
  for (int o = 128; o > 0; o >>= 1) { if (tid < o) red[tid] += red[tid + o]; __syncthreads(); }
  float inv = 1.f / fmaxf(sqrtf(red[0]), 1e-6f);
  float* y = out + (size_t)row * D;
  for (int i = tid; i < D; i += 256) y[i] = x[i] * inv;
}

__global__ __launch_bounds__(256) void row_softmax_kernel(float* __restrict__ W)
{
  __shared__ float red[256];
  int row = blockIdx.x, tid = threadIdx.x;
  float* x = W + (size_t)row * 1024;
  float m = -3.0e38f;
  for (int c = tid; c < CC; c += 256) m = fmaxf(m, x[c]);
  red[tid] = m; __syncthreads();
  for (int o = 128; o > 0; o >>= 1) { if (tid < o) red[tid] = fmaxf(red[tid], red[tid + o]); __syncthreads(); }
  m = red[0];
  __syncthreads();
  float s = 0.f;
  for (int c = tid; c < CC; c += 256) s += expf(x[c] - m);
  red[tid] = s; __syncthreads();
  for (int o = 128; o > 0; o >>= 1) { if (tid < o) red[tid] += red[tid + o]; __syncthreads(); }
  float inv = 1.f / red[0];
  for (int c = tid; c < 1024; c += 256)
    x[c] = (c < CC) ? expf(x[c] - m) * inv : 0.f;
}

__global__ __launch_bounds__(256) void mean_t_kernel(
    const float* __restrict__ vis, float* __restrict__ h)
{
  int idx = blockIdx.x * 256 + threadIdx.x;
  if (idx >= BBATCH * DV) return;
  int b = idx / DV, d = idx - b * DV;
  float s = 0.f;
  const float* p = vis + (size_t)b * TT * DV + d;
  for (int t = 0; t < TT; ++t) s += p[(size_t)t * DV];
  h[idx] = s * (1.f / (float)TT);
}

__global__ __launch_bounds__(256) void prefix_kernel(
    const float* __restrict__ gen,   // (B, P*R)
    const float* __restrict__ Wo,    // (DV, R)
    const float* __restrict__ pre,   // (P, DV)
    float* __restrict__ out)         // (B, P, DV)
{
  int idx = blockIdx.x * 256 + threadIdx.x;
  if (idx >= BBATCH * PFX * DV) return;
  int d = idx % DV;
  int p = (idx / DV) % PFX;
  int b = idx / (DV * PFX);
  float acc = pre[p * DV + d];
  const float* g = gen + (size_t)b * (PFX * RRK) + p * RRK;
  const float* w = Wo + (size_t)d * RRK;
  for (int r2 = 0; r2 < RRK; ++r2) acc += g[r2] * w[r2];
  out[idx] = acc;
}

__global__ __launch_bounds__(256) void fuse_kernel(
    const float* __restrict__ a, const float* __restrict__ x,
    const float* __restrict__ m, float* __restrict__ o, int n)
{
  int i = blockIdx.x * 256 + threadIdx.x;
  if (i < n) o[i] = a[i] + x[i] * m[i];
}

// ---------------------------------------------------------------------------
// Host orchestration
// ---------------------------------------------------------------------------
static inline void gemm(const float* A, const float* Bp, const float* bias,
                        float* Cp, int M, int N, int K, int tb, int K_B,
                        int ldc, int epi, const float* ap, int am,
                        hipStream_t s)
{
  dim3 g((unsigned)((N + 127) / 128), (unsigned)((M + 127) / 128));
  gemm_wmma<<<g, 256, 0, s>>>(A, Bp, bias, Cp, M, N, K, tb, K_B, ldc, epi, ap, am);
}

extern "C" void kernel_launch(void* const* d_in, const int* in_sizes, int n_in,
                              void* d_out, int out_size, void* d_ws, size_t ws_size,
                              hipStream_t stream)
{
  (void)in_sizes; (void)n_in; (void)out_size; (void)ws_size;
  const float* vis      = (const float*)d_in[0];
  const float* txt      = (const float*)d_in[1];
  const float* Wk       = (const float*)d_in[2];
  const float* bk       = (const float*)d_in[3];
  const float* Wv       = (const float*)d_in[4];
  const float* bv       = (const float*)d_in[5];
  const float* lnq_g    = (const float*)d_in[6];
  const float* lnq_b    = (const float*)d_in[7];
  const float* lnkv_g   = (const float*)d_in[8];
  const float* lnkv_b   = (const float*)d_in[9];
  const float* lnout_g  = (const float*)d_in[10];
  const float* lnout_b  = (const float*)d_in[11];
  const float* pool_g   = (const float*)d_in[12];
  const float* pool_b   = (const float*)d_in[13];
  const float* prefix_k = (const float*)d_in[14];
  const float* prefix_v = (const float*)d_in[15];
  const float* Wgk      = (const float*)d_in[16];
  const float* bgk      = (const float*)d_in[17];
  const float* Wgv      = (const float*)d_in[18];
  const float* bgv      = (const float*)d_in[19];
  const float* Wok      = (const float*)d_in[20];
  const float* Wov      = (const float*)d_in[21];
  const float* in_w     = (const float*)d_in[22];
  const float* in_b     = (const float*)d_in[23];
  const float* out_w    = (const float*)d_in[24];
  const float* out_b    = (const float*)d_in[25];
  const float* Wd       = (const float*)d_in[26];
  const float* bd       = (const float*)d_in[27];
  const float* Wu       = (const float*)d_in[28];
  const float* bu       = (const float*)d_in[29];
  const float* Wm       = (const float*)d_in[30];
  const float* bm       = (const float*)d_in[31];
  const float* Wf       = (const float*)d_in[32];
  const float* bf       = (const float*)d_in[33];
  const float* logit_scale = (const float*)d_in[34];

  float* ws = (float*)d_ws;
  size_t off = 0;
  auto take = [&](size_t n) -> float* { float* p = ws + off; off += n; return p; };
  const size_t NVD = (size_t)NVROWS * DV;
  float* bufA  = take(NVD);                 // Q -> F_att -> fused_n
  float* bufB  = take(NVD);                 // qproj -> Qn -> XWf
  float* bufC  = take(NVD);                 // F_att_o -> F_mod
  float* bufD  = take(NVD);                 // F_hat
  float* bufE  = take(NVD);                 // F_txt_ctx -> X
  float* bufG  = take((size_t)NVROWS * BNK);// adapter hidden
  float* wbuf  = take((size_t)NVROWS * 1024);// cosine probs (K padded to 1024)
  float* kbase = take((size_t)CC * DV);
  float* vbase = take((size_t)CC * DV);
  float* kn    = take((size_t)CC * DV);
  float* ktext = take((size_t)CC * DV);
  float* vtext = take((size_t)CC * DV);
  float* hbuf  = take((size_t)BBATCH * DV);
  float* pkr   = take((size_t)BBATCH * BNK);
  float* pvr   = take((size_t)BBATCH * BNK);
  float* Pk    = take((size_t)BBATCH * PFX * DV);
  float* Pv    = take((size_t)BBATCH * PFX * DV);
  float* kpre  = take((size_t)BBATCH * PFX * DV);
  float* vpre  = take((size_t)BBATCH * PFX * DV);

  const float* Wq_ = in_w;
  const float* Wk_ = in_w + (size_t)DV * DV;
  const float* Wv_ = in_w + 2 * (size_t)DV * DV;
  const float* bq_ = in_b;
  const float* bk_ = in_b + DV;
  const float* bv_ = in_b + 2 * DV;

  // 1-2. text K/V bases, LN in place
  gemm(txt, Wk, bk, kbase, CC, DV, DTXT, 1, 0, DV, 0, nullptr, 0, stream);
  gemm(txt, Wv, bv, vbase, CC, DV, DTXT, 1, 0, DV, 0, nullptr, 0, stream);
  ln_kernel<<<CC, 256, 0, stream>>>(kbase, kbase, lnkv_g, lnkv_b, DV);
  ln_kernel<<<CC, 256, 0, stream>>>(vbase, vbase, lnkv_g, lnkv_b, DV);
  // 3. Kn == Tproj
  l2norm_kernel<<<CC, 256, 0, stream>>>(kbase, kn, DV);
  // 4. pooled visual -> LN
  mean_t_kernel<<<(BBATCH * DV + 255) / 256, 256, 0, stream>>>(vis, hbuf);
  ln_kernel<<<BBATCH, 256, 0, stream>>>(hbuf, hbuf, pool_g, pool_b, DV);
  // 5. dynamic prefix
  gemm(hbuf, Wgk, bgk, pkr, BBATCH, BNK, DV, 1, 0, BNK, 0, nullptr, 0, stream);
  gemm(hbuf, Wgv, bgv, pvr, BBATCH, BNK, DV, 1, 0, BNK, 0, nullptr, 0, stream);
  prefix_kernel<<<(BBATCH * PFX * DV + 255) / 256, 256, 0, stream>>>(pkr, Wok, prefix_k, Pk);
  prefix_kernel<<<(BBATCH * PFX * DV + 255) / 256, 256, 0, stream>>>(pvr, Wov, prefix_v, Pv);
  // 6-7. Q = ln(vis); qproj
  ln_kernel<<<NVROWS, 256, 0, stream>>>(vis, bufA, lnq_g, lnq_b, DV);
  gemm(bufA, Wq_, bq_, bufB, NVROWS, DV, DV, 1, 0, DV, 0, nullptr, 0, stream);
  // 8-9. shared text k/v projections (broadcast exploited) + prefix k/v
  gemm(kbase, Wk_, bk_, ktext, CC, DV, DV, 1, 0, DV, 0, nullptr, 0, stream);
  gemm(vbase, Wv_, bv_, vtext, CC, DV, DV, 1, 0, DV, 0, nullptr, 0, stream);
  gemm(Pk, Wk_, bk_, kpre, BBATCH * PFX, DV, DV, 1, 0, DV, 0, nullptr, 0, stream);
  gemm(Pv, Wv_, bv_, vpre, BBATCH * PFX, DV, DV, 1, 0, DV, 0, nullptr, 0, stream);
  // 10. fused attention -> F_att (bufA, Q dead)
  mha_kernel<<<dim3(BBATCH * NH, (TT + 31) / 32), 256, 0, stream>>>(
      bufB, kpre, vpre, ktext, vtext, bufA);
  // 11. output projection
  gemm(bufA, out_w, out_b, bufC, NVROWS, DV, DV, 1, 0, DV, 0, nullptr, 0, stream);
  // 12. bottleneck adapter (qgelu)
  gemm(bufC, Wd, bd, bufG, NVROWS, BNK, DV, 1, 0, BNK, 1, nullptr, 0, stream);
  gemm(bufG, Wu, bu, bufD, NVROWS, DV, BNK, 1, 0, DV, 0, nullptr, 0, stream);
  // 13. cosine-sim attention over text
  l2norm_kernel<<<NVROWS, 256, 0, stream>>>(vis, bufB, DV);             // Qn
  gemm(bufB, kn, nullptr, wbuf, NVROWS, CC, DV, 1, 0, 1024, 0, nullptr, 0, stream);
  row_softmax_kernel<<<NVROWS, 256, 0, stream>>>(wbuf);
  gemm(wbuf, vbase, nullptr, bufE, NVROWS, DV, 1024, 0, CC, DV, 0, nullptr, 0, stream);
  // 14. F_mod = sigmoid(ln(ctx) @ Wm^T + bm)
  ln_kernel<<<NVROWS, 256, 0, stream>>>(bufE, bufE, lnkv_g, lnkv_b, DV);
  gemm(bufE, Wm, bm, bufC, NVROWS, DV, DV, 1, 0, DV, 2, nullptr, 0, stream);
  // 15-16. fuse, Wf, LN -> F_fused (second output)
  fuse_kernel<<<(int)((NVD + 255) / 256), 256, 0, stream>>>(vis, bufD, bufC, bufE, (int)NVD);
  gemm(bufE, Wf, bf, bufB, NVROWS, DV, DV, 1, 0, DV, 0, nullptr, 0, stream);
  float* f_out = (float*)d_out + (size_t)NVROWS * CC;
  ln_kernel<<<NVROWS, 256, 0, stream>>>(bufB, f_out, lnout_g, lnout_b, DV);
  // 17-18. logits = l2norm(F_fused) @ Kn^T * exp(min(logit_scale, ln 100))
  l2norm_kernel<<<NVROWS, 256, 0, stream>>>(f_out, bufA, DV);
  gemm(bufA, kn, nullptr, (float*)d_out, NVROWS, CC, DV, 1, 0, CC, 0,
       logit_scale, 1, stream);
}
